// SpatialGatingUnit_Sigmoid_9826885173358
// MI455X (gfx1250) — matrix operand order repaired and checked by reading, not compiled, not run on hardware
//
#include <hip/hip_runtime.h>

#define DD 2048
#define SS 512
#define BB 32

#if __has_builtin(__builtin_amdgcn_global_load_async_to_lds_b128)
#define HAS_ASYNC 1
#else
#define HAS_ASYNC 0
#endif

#if __has_builtin(__builtin_amdgcn_s_wait_asynccnt)
#define WAIT_ASYNC() __builtin_amdgcn_s_wait_asynccnt(0)
#else
#define WAIT_ASYNC() asm volatile("s_wait_asynccnt 0x0" ::: "memory")
#endif

typedef __attribute__((ext_vector_type(16))) __bf16 v16bf;
typedef __attribute__((ext_vector_type(8)))  __bf16 v8bf;
typedef __attribute__((ext_vector_type(8)))  float  v8f;
typedef int v4i __attribute__((vector_size(16)));
typedef v4i __attribute__((address_space(1)))* gas_v4i;   // global (AS1) int4*
typedef v4i __attribute__((address_space(3)))* las_v4i;   // LDS    (AS3) int4*

__device__ __forceinline__ unsigned short f2bf(float f) {
  unsigned u = __float_as_uint(f);
  u += 0x7FFFu + ((u >> 16) & 1u);   // round-to-nearest-even
  return (unsigned short)(u >> 16);
}
__device__ __forceinline__ float bf2f(unsigned short h) {
  return __uint_as_float(((unsigned)h) << 16);
}
__device__ __forceinline__ float gelu_exact(float v) {
  return 0.5f * v * (1.0f + erff(v * 0.70710678118654752440f));
}

// ---------------- Kernel 1: LayerNorm -> bf16 xn ----------------
__global__ __launch_bounds__(256) void ln_kernel(const float* __restrict__ x,
                                                 const float* __restrict__ lnw,
                                                 const float* __restrict__ lnb,
                                                 unsigned short* __restrict__ xnbf) {
  const int row = blockIdx.x;            // b*S + s  (16384 rows)
  const int t   = threadIdx.x;           // 256 threads, 8 elems each
  const float* xr = x + (size_t)row * DD;

  float4 v0 = *(const float4*)(xr + t * 8);
  float4 v1 = *(const float4*)(xr + t * 8 + 4);
  float s  = v0.x + v0.y + v0.z + v0.w + v1.x + v1.y + v1.z + v1.w;
  float ss = v0.x*v0.x + v0.y*v0.y + v0.z*v0.z + v0.w*v0.w
           + v1.x*v1.x + v1.y*v1.y + v1.z*v1.z + v1.w*v1.w;

  #pragma unroll
  for (int o = 16; o > 0; o >>= 1) {
    s  += __shfl_xor(s, o, 32);
    ss += __shfl_xor(ss, o, 32);
  }
  __shared__ float rs[8], rq[8];
  const int wave = t >> 5, lane = t & 31;
  if (lane == 0) { rs[wave] = s; rq[wave] = ss; }
  __syncthreads();
  float S = 0.f, Q = 0.f;
  #pragma unroll
  for (int i = 0; i < 8; ++i) { S += rs[i]; Q += rq[i]; }

  const float mu   = S * (1.0f / DD);
  const float var  = Q * (1.0f / DD) - mu * mu;
  const float rstd = rsqrtf(var + 1e-5f);

  float4 w0 = *(const float4*)(lnw + t * 8);
  float4 w1 = *(const float4*)(lnw + t * 8 + 4);
  float4 b0 = *(const float4*)(lnb + t * 8);
  float4 b1 = *(const float4*)(lnb + t * 8 + 4);

  alignas(16) unsigned short o[8];
  o[0] = f2bf((v0.x - mu) * rstd * w0.x + b0.x);
  o[1] = f2bf((v0.y - mu) * rstd * w0.y + b0.y);
  o[2] = f2bf((v0.z - mu) * rstd * w0.z + b0.z);
  o[3] = f2bf((v0.w - mu) * rstd * w0.w + b0.w);
  o[4] = f2bf((v1.x - mu) * rstd * w1.x + b1.x);
  o[5] = f2bf((v1.y - mu) * rstd * w1.y + b1.y);
  o[6] = f2bf((v1.z - mu) * rstd * w1.z + b1.z);
  o[7] = f2bf((v1.w - mu) * rstd * w1.w + b1.w);
  *(uint4*)(xnbf + (size_t)row * DD + t * 8) = *(const uint4*)o;
}

// ---------------- Kernel 2: wi/wii -> bf16 ----------------
__global__ __launch_bounds__(256) void convw_kernel(const float* __restrict__ wi,
                                                    const float* __restrict__ wii,
                                                    unsigned short* __restrict__ wbf) {
  const int i = (blockIdx.x * 256 + threadIdx.x) * 4;   // 262144 elems per matrix
  float4 a = *(const float4*)(wi + i);
  float4 b = *(const float4*)(wii + i);
  alignas(8) unsigned short oa[4] = { f2bf(a.x), f2bf(a.y), f2bf(a.z), f2bf(a.w) };
  alignas(8) unsigned short ob[4] = { f2bf(b.x), f2bf(b.y), f2bf(b.z), f2bf(b.w) };
  *(uint2*)(wbf + i)           = *(const uint2*)oa;
  *(uint2*)(wbf + SS * SS + i) = *(const uint2*)ob;
}

// ---------------- Kernel 3: gating MLP -> idx[b] ----------------
__global__ __launch_bounds__(256) void gate_kernel(const unsigned short* __restrict__ xnbf,
                                                   const float* __restrict__ w1,
                                                   const float* __restrict__ b1,
                                                   const float* __restrict__ w2,
                                                   const float* __restrict__ b2,
                                                   int* __restrict__ idxw) {
  const int b = blockIdx.x, t = threadIdx.x;
  const unsigned short* cls = xnbf + (size_t)b * SS * DD;   // row s = 0
  float h[10];
  #pragma unroll
  for (int j = 0; j < 10; ++j) h[j] = 0.f;
  for (int d = t; d < DD; d += 256) {
    const float c = bf2f(cls[d]);
    const float* wr = w1 + d * 10;
    #pragma unroll
    for (int j = 0; j < 10; ++j) h[j] += c * wr[j];
  }
  __shared__ float red[10][256];
  #pragma unroll
  for (int j = 0; j < 10; ++j) red[j][t] = h[j];
  __syncthreads();
  for (int off = 128; off > 0; off >>= 1) {
    if (t < off) {
      #pragma unroll
      for (int j = 0; j < 10; ++j) red[j][t] += red[j][t + off];
    }
    __syncthreads();
  }
  if (t == 0) {
    float z = b2[0];
    #pragma unroll
    for (int j = 0; j < 10; ++j) z += gelu_exact(red[j][0] + b1[j]) * w2[j];
    idxw[b] = (z > 0.f) ? 1 : 0;   // round(sigmoid(z)) == (z > 0)
  }
}

// ---------------- Kernel 4: batched WMMA GEMM + bias + residual + GELU ----------------
// out[b] = gelu( W[idx[b]] @ xn[b] + bias[idx[b]] + x[b] )
// block tile: 128(M=s) x 128(N=d), double-buffered K=32 steps
// 8 waves; wave tile 32x64 -> 2 A-frags x 4 B-frags = 8 WMMA per K-step
__global__ __launch_bounds__(256) void gemm_kernel(const unsigned short* __restrict__ xnbf,
                                                   const unsigned short* __restrict__ wbf,
                                                   const int* __restrict__ idxw,
                                                   const float* __restrict__ x,
                                                   const float* __restrict__ bi,
                                                   const float* __restrict__ bii,
                                                   float* __restrict__ out) {
  __shared__ unsigned short Alds[2][128][40];   // [buf][M][K], stride 40 keeps 16B align
  __shared__ unsigned short Blds[2][128][40];   // [buf][N][K]  (transposed xn tile)

  const int b    = blockIdx.z;
  const int s0   = blockIdx.y * 128;
  const int n0   = blockIdx.x * 128;
  const int t    = threadIdx.x;
  const int lane = t & 31, wave = t >> 5;
  const int wm   = wave & 3;    // M strip: 32 rows each
  const int wn   = wave >> 2;   // N half: 64 cols each
  const int hl   = lane & 15, hi = lane >> 4;

  const int idxb = idxw[b];
  const unsigned short* Wp = wbf + (size_t)idxb * (SS * SS);
  const unsigned short* Xp = xnbf + (size_t)b * SS * DD;

  // staging maps (256 threads)
  const int arow = t >> 1;            // 0..127  (A: 128 x 32, two 16B chunks/thread)
  const int acol = (t & 1) * 16;      // 0 | 16
  const int brow = t >> 3;            // 0..31   (B: 32 x 128, 16 halfs/thread)
  const int bcol = (t & 7) * 16;

  v8f acc[2][4] = {};
  uint4 bq0, bq1;
#if !HAS_ASYNC
  uint4 aq0, aq1;
#endif

  // ---- prologue: stage K-step 0 ----
  {
    const unsigned short* xb = Xp + (size_t)brow * DD + n0 + bcol;
    bq0 = *(const uint4*)xb;
    bq1 = *(const uint4*)(xb + 8);
    const unsigned short* ga = Wp + (size_t)(s0 + arow) * SS + acol;
#if HAS_ASYNC
    __builtin_amdgcn_global_load_async_to_lds_b128(
        (gas_v4i)ga,       (las_v4i)&Alds[0][arow][acol],     0, 0);
    __builtin_amdgcn_global_load_async_to_lds_b128(
        (gas_v4i)(ga + 8), (las_v4i)&Alds[0][arow][acol + 8], 0, 0);
#else
    aq0 = *(const uint4*)ga;
    aq1 = *(const uint4*)(ga + 8);
#endif
  }

  for (int it = 0; it < 16; ++it) {
    const int cur = it & 1;

    // finish staging buffer 'cur': B transpose-scatter (+ A store on fallback path)
    {
      alignas(16) unsigned short e[16];
      *(uint4*)(e)     = bq0;
      *(uint4*)(e + 8) = bq1;
      #pragma unroll
      for (int i = 0; i < 16; ++i) Blds[cur][bcol + i][brow] = e[i];
#if !HAS_ASYNC
      *(uint4*)&Alds[cur][arow][acol]     = aq0;
      *(uint4*)&Alds[cur][arow][acol + 8] = aq1;
#endif
    }
#if HAS_ASYNC
    WAIT_ASYNC();
#endif
    __syncthreads();

    // prefetch next K-step into the other buffer (overlaps with compute below)
    if (it < 15) {
      const int kk = (it + 1) * 32;
      const unsigned short* xb = Xp + (size_t)(kk + brow) * DD + n0 + bcol;
      bq0 = *(const uint4*)xb;
      bq1 = *(const uint4*)(xb + 8);
      const unsigned short* ga = Wp + (size_t)(s0 + arow) * SS + kk + acol;
#if HAS_ASYNC
      __builtin_amdgcn_global_load_async_to_lds_b128(
          (gas_v4i)ga,       (las_v4i)&Alds[1 - cur][arow][acol],     0, 0);
      __builtin_amdgcn_global_load_async_to_lds_b128(
          (gas_v4i)(ga + 8), (las_v4i)&Alds[1 - cur][arow][acol + 8], 0, 0);
#else
      aq0 = *(const uint4*)ga;
      aq1 = *(const uint4*)(ga + 8);
#endif
    }

    // compute on buffer 'cur'
    // A fragment (16x32 bf16): lanes 0-15: K 0-7/16-23 ; lanes 16-31: K 8-15/24-31
    const int akb = hi ? 8 : 0;
    v16bf af[2];
    #pragma unroll
    for (int sIdx = 0; sIdx < 2; ++sIdx) {
      const int r = wm * 32 + sIdx * 16 + hl;
      ((v8bf*)&af[sIdx])[0] = *(const v8bf*)&Alds[cur][r][akb];
      ((v8bf*)&af[sIdx])[1] = *(const v8bf*)&Alds[cur][r][akb + 16];
    }
    // B fragment (32x16 bf16): lane = N col; lanes 0-15: K 0-7/16-23 ; lanes 16-31: K 8-15/24-31
    const int bkb = hi ? 8 : 0;
    v16bf bf[4];
    #pragma unroll
    for (int j = 0; j < 4; ++j) {
      const int r = wn * 64 + j * 16 + hl;
      ((v8bf*)&bf[j])[0] = *(const v8bf*)&Blds[cur][r][bkb];
      ((v8bf*)&bf[j])[1] = *(const v8bf*)&Blds[cur][r][bkb + 16];
    }
    #pragma unroll
    for (int sIdx = 0; sIdx < 2; ++sIdx) {
      #pragma unroll
      for (int j = 0; j < 4; ++j) {
        acc[sIdx][j] = __builtin_amdgcn_wmma_f32_16x16x32_bf16(
            false, af[sIdx], false, bf[j], (short)0, acc[sIdx][j], false, false);
      }
    }
    // no trailing barrier needed: next iteration writes the opposite buffer and
    // begins reading this one only after its top barrier.
  }

  // epilogue: C layout VGPR r -> (M = r + 8*hi, N = hl)
  const float* bptr = idxb ? bii : bi;
  #pragma unroll
  for (int sIdx = 0; sIdx < 2; ++sIdx) {
    float biasr[8];
    #pragma unroll
    for (int r = 0; r < 8; ++r) biasr[r] = bptr[s0 + wm * 32 + sIdx * 16 + hi * 8 + r];
    #pragma unroll
    for (int j = 0; j < 4; ++j) {
      const int dcol = n0 + wn * 64 + j * 16 + hl;
      #pragma unroll
      for (int r = 0; r < 8; ++r) {
        const int srow = s0 + wm * 32 + sIdx * 16 + hi * 8 + r;
        const size_t off = ((size_t)b * SS + srow) * DD + dcol;
        out[off] = gelu_exact(acc[sIdx][j][r] + biasr[r] + x[off]);
      }
    }
  }
}

extern "C" void kernel_launch(void* const* d_in, const int* in_sizes, int n_in,
                              void* d_out, int out_size, void* d_ws, size_t ws_size,
                              hipStream_t stream) {
  (void)in_sizes; (void)n_in; (void)out_size; (void)ws_size;
  const float* x   = (const float*)d_in[0];
  const float* lnw = (const float*)d_in[1];
  const float* lnb = (const float*)d_in[2];
  const float* w1  = (const float*)d_in[3];
  const float* b1  = (const float*)d_in[4];
  const float* w2  = (const float*)d_in[5];
  const float* b2  = (const float*)d_in[6];
  const float* wi  = (const float*)d_in[7];
  const float* bi  = (const float*)d_in[8];
  const float* wii = (const float*)d_in[9];
  const float* bii = (const float*)d_in[10];
  float* out = (float*)d_out;

  unsigned short* xnbf = (unsigned short*)d_ws;                             // 64 MB
  unsigned short* wbf  = (unsigned short*)((char*)d_ws + (size_t)67108864); // 1 MB
  int*            idxw = (int*)((char*)d_ws + (size_t)67108864 + 1048576);  // 128 B

  ln_kernel   <<<dim3(BB * SS), dim3(256), 0, stream>>>(x, lnw, lnb, xnbf);
  convw_kernel<<<dim3(256),     dim3(256), 0, stream>>>(wi, wii, wbf);
  gate_kernel <<<dim3(BB),      dim3(256), 0, stream>>>(xnbf, w1, b1, w2, b2, idxw);
  gemm_kernel <<<dim3(DD / 128, SS / 128, BB), dim3(256), 0, stream>>>(
      xnbf, wbf, idxw, x, bi, bii, out);
}
